// AnchorSCNN_48284022341785
// MI455X (gfx1250) — matrix operand order repaired and checked by reading, not compiled, not run on hardware
//
#include <hip/hip_runtime.h>
#include <math.h>

typedef __attribute__((ext_vector_type(16))) _Float16 v16h;
typedef __attribute__((ext_vector_type(8)))  float    v8f;

#define GVOX 32
#define NPTS 200000

// ---------------------------------------------------------------------------
// Implicit-GEMM conv3d / point-MLP on V_WMMA_F32_16X16X32_F16.
// One wave -> 16 rows (voxels/points) x 16 out-channels tile.
// Fully templated: <TA activation type, TO output type, KS, CIN, COUT,
// GIN, GOUT, STRIDE, PAD>. All geometry math is compile-time (mul/shift only),
// the hot loop is contiguous loads + wmma (A fragments are 16 contiguous k per
// lane, B fragments 16 contiguous halves from pre-transposed Wt[Cout][K]).
// Invalid taps read a zero row instead of predicating the loads.
// Point-MLP mode: KS=1, STRIDE=1, PAD=0, GIN=GOUT=G with G^3 >= rows (base-G
// digit recomposition collapses the index to row*CIN + ci).
// ---------------------------------------------------------------------------
template <typename TA, typename TO, int KS, int CIN, int COUT,
          int GIN, int GOUT, int STRIDE, int PAD>
__global__ __launch_bounds__(32)
void conv_wmma(const TA* __restrict__ X, const _Float16* __restrict__ Wt,
               const float* __restrict__ lbias,
               const float* __restrict__ bng, const float* __restrict__ bnb,
               const float* __restrict__ bnm, const float* __restrict__ bnv,
               const float* __restrict__ mask, const float* __restrict__ res,
               TO* __restrict__ Y, const TA* __restrict__ zrow,
               int relu_flag, int accum_flag) {
  constexpr int KS3 = KS * KS * KS;
  const int lane = threadIdx.x & 31;
  const int hid  = lane >> 4;           // lane half (0/1)
  const int lr   = lane & 15;
  const int m0   = blockIdx.x << 4;
  const int n0   = blockIdx.y << 4;
  const int col  = n0 + lr;
  const bool colOK = (col < COUT);      // folds away when COUT%16==0
  const int rowA = m0 + lr;
  const int vz = rowA / (GOUT * GOUT);  // constexpr divisor -> mul/shift
  const int rr = rowA - vz * (GOUT * GOUT);
  const int vy = rr / GOUT;
  const int vx = rr - vy * GOUT;
  const int bz = vz * STRIDE - PAD;
  const int by = vy * STRIDE - PAD;
  const int bx = vx * STRIDE - PAD;
  const _Float16* __restrict__ wrow = Wt + (size_t)(colOK ? col : 0) * (KS3 * CIN);
  __builtin_prefetch(wrow, 0, 1);

  v8f acc = {0.f, 0.f, 0.f, 0.f, 0.f, 0.f, 0.f, 0.f};
  for (int tap = 0; tap < KS3; ++tap) {
    const int dz = tap / (KS * KS);
    const int rt = tap - dz * (KS * KS);
    const int dy = rt / KS;
    const int dx = rt - dy * KS;
    const int iz = bz + dz, iy = by + dy, ix = bx + dx;
    const bool valid = (unsigned)iz < (unsigned)GIN &&
                       (unsigned)iy < (unsigned)GIN &&
                       (unsigned)ix < (unsigned)GIN;
    const TA* __restrict__ Xp =
        valid ? (X + (((size_t)iz * GIN + iy) * GIN + ix) * CIN) : zrow;
    const _Float16* __restrict__ Wp = wrow + tap * CIN;
#pragma unroll
    for (int cb = 0; cb < CIN; cb += 32) {
      v16h av, bv;
#pragma unroll
      for (int j = 0; j < 16; ++j) {
        // A (16x32 f16): lanes 0-15 -> K 0..15, lanes 16-31 -> K 16..31
        const int ciA = cb + (hid << 4) + j;
        // B (32x16 f16): lanes 0-15 -> K 0..15, lanes 16-31 -> K 16..31
        const int ciB = cb + (hid << 4) + j;
        av[j] = (ciA < CIN) ? (_Float16)Xp[ciA] : (_Float16)0.0f;   // guard folds
        bv[j] = (ciB < CIN) ? Wp[ciB] : (_Float16)0.0f;             // guard folds
      }
      acc = __builtin_amdgcn_wmma_f32_16x16x32_f16(false, av, false, bv,
                                                   (short)0, acc, false, false);
    }
  }

  const int cc = colOK ? col : 0;
  const float sc = bng[cc] * rsqrtf(bnv[cc] + 1e-5f);
  const float bb = bnb[cc];
  const float mm = bnm[cc];
  const float bs = lbias ? lbias[cc] : 0.0f;
#pragma unroll
  for (int r = 0; r < 8; ++r) {
    const int row = m0 + (hid << 3) + r;          // D: row = r + 8*half, col = lane%16
    float val = acc[r] + bs;
    val = (val - mm) * sc + bb;
    if (res) val += res[(size_t)row * COUT + cc];
    if (relu_flag) val = fmaxf(val, 0.0f);
    if (mask) val *= mask[row];
    if (colOK) {
      const size_t o = (size_t)row * COUT + col;
      if (accum_flag) Y[o] = (TO)((float)Y[o] + val);
      else            Y[o] = (TO)val;
    }
  }
}

// Transpose + f16-convert weights: W[K][Cout] (fp32) -> Wt[Cout][K] (f16)
__global__ void k_prep_w(const float* __restrict__ W, _Float16* __restrict__ Wt,
                         int K, int Cout) {
  int t = blockIdx.x * blockDim.x + threadIdx.x;
  if (t >= K * Cout) return;
  int k = t / Cout, co = t - k * Cout;
  Wt[(size_t)co * K + k] = (_Float16)W[t];
}

// ---------------------------------------------------------------------------
// Scalar helper kernels
// ---------------------------------------------------------------------------
__device__ __forceinline__ void point_cell(const float* __restrict__ pts,
                                           const int* pres, const int* vres,
                                           int n, int& i0, int& i1, int& i2) {
  float scale = (float)pres[0] / (float)vres[0];
  float s = (32.0f - 1e-3f) * scale;
  i0 = (int)floorf(pts[3 * n + 0] * s);
  i1 = (int)floorf(pts[3 * n + 1] * s);
  i2 = (int)floorf(pts[3 * n + 2] * s);
  i0 = min(max(i0, 0), GVOX - 1);
  i1 = min(max(i1, 0), GVOX - 1);
  i2 = min(max(i2, 0), GVOX - 1);
}

__global__ void k_scatter_base(const float* __restrict__ pts, const float* __restrict__ zd,
                               const int* pres, const int* vres,
                               float* cnt1, float* vdsum, int N) {
  int n = blockIdx.x * blockDim.x + threadIdx.x;
  if (n >= N) return;
  int i0, i1, i2;
  point_cell(pts, pres, vres, n, i0, i1, i2);
  int lin = (i0 * GVOX + i1) * GVOX + i2;
  atomicAdd(&cnt1[lin], 1.0f);
  atomicAdd(&vdsum[lin], zd[n]);
}

__global__ void k_finalize_base(const float* __restrict__ cnt1, float* vd, float* mask0, int V) {
  int v = blockIdx.x * blockDim.x + threadIdx.x;
  if (v >= V) return;
  float c = cnt1[v];
  vd[v] = vd[v] / fmaxf(c, 1.0f);
  mask0[v] = (c > 0.0f) ? 1.0f : 0.0f;
}

__global__ void k_pool(const float* __restrict__ mask0, const float* __restrict__ cnt1,
                       float* mask1, float* cnt2) {
  int v = blockIdx.x * blockDim.x + threadIdx.x;
  if (v >= 16 * 16 * 16) return;
  int z = v >> 8, y = (v >> 4) & 15, x = v & 15;
  float mx = 0.0f, sm = 0.0f;
#pragma unroll
  for (int d = 0; d < 8; ++d) {
    int cz = 2 * z + (d >> 2), cy = 2 * y + ((d >> 1) & 1), cx = 2 * x + (d & 1);
    int c = (cz * GVOX + cy) * GVOX + cx;
    mx = fmaxf(mx, mask0[c]);
    sm += cnt1[c];
  }
  mask1[v] = mx;
  cnt2[v] = sm;
}

// feat_grid * mask0 -> f16 (feeds WMMA stem only)
__global__ void k_maskmul_h(const float* __restrict__ src, const float* __restrict__ mask0,
                            _Float16* __restrict__ dst, int V, int C) {
  int t = blockIdx.x * blockDim.x + threadIdx.x;
  if (t >= V * C) return;
  dst[t] = (_Float16)(src[t] * mask0[t / C]);
}

// stem_d: 1-channel 3^3 stencil -> 48ch, CBR, masked (scalar; WMMA would be 1/32 full)
__global__ void k_stem_d(const float* __restrict__ vd, const float* __restrict__ w,
                         const float* __restrict__ g, const float* __restrict__ b,
                         const float* __restrict__ m, const float* __restrict__ v,
                         const float* __restrict__ mask0, float* __restrict__ x0) {
  int t = blockIdx.x * blockDim.x + threadIdx.x;
  if (t >= GVOX * GVOX * GVOX * 48) return;
  int vox = t / 48, co = t - vox * 48;
  int z = vox >> 10, y = (vox >> 5) & 31, x = vox & 31;
  float a = 0.0f;
#pragma unroll
  for (int tap = 0; tap < 27; ++tap) {
    int dz = tap / 9, rt = tap - dz * 9, dy = rt / 3, dx = rt - dy * 3;
    int iz = z + dz - 1, iy = y + dy - 1, ix = x + dx - 1;
    if ((unsigned)iz < 32u && (unsigned)iy < 32u && (unsigned)ix < 32u)
      a += vd[(iz * GVOX + iy) * GVOX + ix] * w[tap * 48 + co];
  }
  a = (a - m[co]) * (g[co] * rsqrtf(v[co] + 1e-5f)) + b[co];
  x0[t] = fmaxf(a, 0.0f) * mask0[vox];
}

__global__ void k_scatter_feat(const float* __restrict__ pts, const float* __restrict__ feats,
                               const int* pres, const int* vres,
                               float* sums, int N, int C, int shift, int Gs) {
  int n = blockIdx.x * blockDim.x + threadIdx.x;
  if (n >= N) return;
  int i0, i1, i2;
  point_cell(pts, pres, vres, n, i0, i1, i2);
  i0 >>= shift; i1 >>= shift; i2 >>= shift;
  size_t lin = ((size_t)(i0 * Gs + i1) * Gs + i2) * C;
  size_t src = (size_t)n * C;
  for (int c = 0; c < C; ++c) atomicAdd(&sums[lin + c], feats[src + c]);
}

// fp32 scatter-mean -> fp32 in place (consumer is scalar)
__global__ void k_norm_mean(float* sums, const float* __restrict__ cnt,
                            const float* __restrict__ mask, int V, int C) {
  int t = blockIdx.x * blockDim.x + threadIdx.x;
  if (t >= V * C) return;
  int v = t / C;
  float val = sums[t] / fmaxf(cnt[v], 1.0f);
  if (mask) val *= mask[v];
  sums[t] = val;
}

// fp32 scatter-mean -> f16 copy (consumer is a WMMA conv)
__global__ void k_norm_mean_h(const float* __restrict__ sums, const float* __restrict__ cnt,
                              const float* __restrict__ mask, _Float16* __restrict__ out,
                              int V, int C) {
  int t = blockIdx.x * blockDim.x + threadIdx.x;
  if (t >= V * C) return;
  int v = t / C;
  out[t] = (_Float16)(sums[t] / fmaxf(cnt[v], 1.0f) * mask[v]);
}

__global__ void k_devox(const float* __restrict__ grid, const float* __restrict__ pts,
                        const int* pres, const int* vres,
                        float* out, int N, int Gs, int C, int stride, int accum) {
  int t = blockIdx.x * blockDim.x + threadIdx.x;
  if (t >= N * C) return;
  int n = t / C, c = t - n * C;
  float scale = (float)pres[0] / (float)vres[0];
  float s = (32.0f - 1e-3f) * scale / (float)stride;
  float q0 = pts[3 * n + 0] * s, q1 = pts[3 * n + 1] * s, q2 = pts[3 * n + 2] * s;
  float g0 = floorf(q0), g1 = floorf(q1), g2 = floorf(q2);
  float f0 = q0 - g0, f1 = q1 - g1, f2 = q2 - g2;
  int i0 = (int)g0, i1 = (int)g1, i2 = (int)g2;
  float acc = 0.0f;
#pragma unroll
  for (int d = 0; d < 8; ++d) {
    int dz = d >> 2, dy = (d >> 1) & 1, dx = d & 1;
    int z = i0 + dz, y = i1 + dy, x = i2 + dx;
    if ((unsigned)z < (unsigned)Gs && (unsigned)y < (unsigned)Gs && (unsigned)x < (unsigned)Gs) {
      float w = (dz ? f0 : 1.0f - f0) * (dy ? f1 : 1.0f - f1) * (dx ? f2 : 1.0f - f2);
      acc += w * grid[(((size_t)z * Gs + y) * Gs + x) * C + c];
    }
  }
  if (accum) out[t] += acc; else out[t] = acc;
}

// conv_transpose 2x2x2 stride 2 'VALID': out[o] = x[o>>1] * w[1-(o&1)]; f16 out -> y2cat
__global__ void k_deconv(const float* __restrict__ in16, const float* __restrict__ w,
                         const float* __restrict__ g, const float* __restrict__ b,
                         const float* __restrict__ m, const float* __restrict__ v,
                         const float* __restrict__ mask0, _Float16* __restrict__ y2cat) {
  int t = blockIdx.x * blockDim.x + threadIdx.x;
  if (t >= GVOX * GVOX * GVOX * 24) return;
  int vox = t / 24, co = t - vox * 24;
  int z = vox >> 10, y = (vox >> 5) & 31, x = vox & 31;
  int src = ((z >> 1) * 16 + (y >> 1)) * 16 + (x >> 1);
  int tz = 1 - (z & 1), ty = 1 - (y & 1), tx = 1 - (x & 1);
  const float* wp = w + ((size_t)((tz * 2 + ty) * 2 + tx) * 64) * 24 + co;
  const float* ip = in16 + (size_t)src * 64;
  float a = 0.0f;
  for (int ci = 0; ci < 64; ++ci) a += ip[ci] * wp[(size_t)ci * 24];
  a = (a - m[co]) * (g[co] * rsqrtf(v[co] + 1e-5f)) + b[co];
  a = fmaxf(a, 0.0f) * mask0[vox];
  y2cat[(size_t)vox * 72 + co] = (_Float16)a;
}

__global__ void k_concat(const float* __restrict__ x0, _Float16* __restrict__ y2cat, int V) {
  int t = blockIdx.x * blockDim.x + threadIdx.x;
  if (t >= V * 48) return;
  int vox = t / 48, c = t - vox * 48;
  y2cat[(size_t)vox * 72 + 24 + c] = (_Float16)x0[t];
}

__global__ void k_occ(const float* __restrict__ z2, const float* __restrict__ w,
                      const float* __restrict__ b, float* __restrict__ out, int N) {
  int n = blockIdx.x * blockDim.x + threadIdx.x;
  if (n >= N) return;
  const float* zp = z2 + (size_t)n * 24;
  float a = 0.0f;
#pragma unroll
  for (int c = 0; c < 24; ++c) a += zp[c] * w[c];
  out[n] = a + b[0];
}

// ---------------------------------------------------------------------------
// Launcher
// ---------------------------------------------------------------------------
static inline int cdiv(int a, int b) { return (a + b - 1) / b; }

extern "C" void kernel_launch(void* const* d_in, const int* in_sizes, int n_in,
                              void* d_out, int out_size, void* d_ws, size_t ws_size,
                              hipStream_t stream) {
  (void)in_sizes; (void)n_in; (void)out_size; (void)ws_size;
  const float* points = (const float*)d_in[0];
  const float* zdepth = (const float*)d_in[1];
  const float* featg  = (const float*)d_in[2];
  const int*   pres   = (const int*)d_in[3];
  const int*   vres   = (const int*)d_in[4];
  auto F = [&](int i) { return (const float*)d_in[i]; };
  // params flattened in setup_inputs() insertion order (bn = g,b,m,v; lin = w,b)
  const float *sd_w = F(5),  *sd_g = F(6),  *sd_b = F(7),  *sd_m = F(8),  *sd_v = F(9);
  const float *sf_w = F(10), *sf_g = F(11), *sf_b = F(12), *sf_m = F(13), *sf_v = F(14);
  const float *dn_w = F(15), *dn_g = F(16), *dn_b = F(17), *dn_m = F(18), *dn_v = F(19);
  const float *r1w1 = F(20), *r1g1 = F(21), *r1b1 = F(22), *r1m1 = F(23), *r1v1 = F(24);
  const float *r1w2 = F(25), *r1g2 = F(26), *r1b2 = F(27), *r1m2 = F(28), *r1v2 = F(29);
  const float *r1wd = F(30), *r1gd = F(31), *r1bd = F(32), *r1md = F(33), *r1vd = F(34);
  const float *dc_w = F(35), *dc_g = F(36), *dc_b = F(37), *dc_m = F(38), *dc_v = F(39);
  const float *u1w1 = F(40), *u1g1 = F(41), *u1b1 = F(42), *u1m1 = F(43), *u1v1 = F(44);
  const float *u1w2 = F(45), *u1g2 = F(46), *u1b2 = F(47), *u1m2 = F(48), *u1v2 = F(49);
  const float *u1wd = F(50), *u1gd = F(51), *u1bd = F(52), *u1md = F(53), *u1vd = F(54);
  const float *p0_w = F(55), *p0_b = F(56), *p0g = F(57), *p0b = F(58), *p0m = F(59), *p0v = F(60);
  const float *p1_w = F(61), *p1_b = F(62), *p1g = F(63), *p1b = F(64), *p1m = F(65), *p1v = F(66);
  const float *oc_w = F(67), *oc_b = F(68);

  const int V32 = GVOX * GVOX * GVOX;     // 32768
  const int V16 = 16 * 16 * 16;           // 4096
  const int N = NPTS;

  float* wsf = (float*)d_ws;
  size_t off = 0;
  auto alloc = [&](size_t n) { float* p = wsf + off; off += (n + 3) & ~(size_t)3; return p; };
  auto allocH = [&](size_t nh) { return (_Float16*)alloc((nh + 1) / 2); };
  float* cnt1  = alloc(V32);
  float* vd    = alloc(V32);
  float* mask0 = alloc(V32);
  float* mask1 = alloc(V16);
  float* cnt2  = alloc(V16);
  float* zrow  = alloc(128);              // zero row (fp32 and f16 views share bits)
  float* x0    = alloc((size_t)V32 * 48);
  float* z0    = alloc((size_t)N * 48);   // reused as z2 (N*24) at the end
  float* g32   = alloc((size_t)V32 * 48); // fp32 scatter-accumulator
  float* s1    = alloc((size_t)V16 * 64);
  float* x1    = alloc((size_t)V16 * 64);
  float* z1    = alloc((size_t)N * 64);
  float* g16   = alloc((size_t)V16 * 64);
  float* s2    = alloc((size_t)V32 * 24);
  float* y2f   = alloc((size_t)V32 * 24);
  float* z2    = z0;                      // alias: z0 dead after ptmlp0
  // f16 activations (consumed only by WMMA convs)
  _Float16* featm_h = allocH((size_t)V32 * 32);
  _Float16* g32h    = allocH((size_t)V32 * 48);
  _Float16* x1a_h   = allocH((size_t)V16 * 48);
  _Float16* h1h     = allocH((size_t)V16 * 64);
  _Float16* y2cat_h = allocH((size_t)V32 * 72);
  _Float16* h2h     = allocH((size_t)V32 * 24);
  // transposed f16 weights  Wt[Cout][K]
  _Float16* t_sf  = allocH(864 * 48);
  _Float16* t_dn  = allocH(384 * 48);
  _Float16* t_r11 = allocH(1296 * 64);
  _Float16* t_r1d = allocH(48 * 64);
  _Float16* t_r12 = allocH(1728 * 64);
  _Float16* t_u11 = allocH(1944 * 24);
  _Float16* t_u1d = allocH(72 * 24);
  _Float16* t_u12 = allocH(648 * 24);
  _Float16* t_p0  = allocH(48 * 64);
  _Float16* t_p1  = allocH(64 * 24);
  const _Float16* zrowh = (const _Float16*)zrow;

  const int B = 256;
  hipMemsetAsync(cnt1, 0, sizeof(float) * V32, stream);
  hipMemsetAsync(vd,   0, sizeof(float) * V32, stream);
  hipMemsetAsync(zrow, 0, sizeof(float) * 128, stream);
  hipMemsetAsync(g32,  0, sizeof(float) * (size_t)V32 * 48, stream);
  hipMemsetAsync(g16,  0, sizeof(float) * (size_t)V16 * 64, stream);

  // 0. weight transpose + f16 conversion
  k_prep_w<<<cdiv(864 * 48, B), B, 0, stream>>>(sf_w, t_sf, 864, 48);
  k_prep_w<<<cdiv(384 * 48, B), B, 0, stream>>>(dn_w, t_dn, 384, 48);
  k_prep_w<<<cdiv(1296 * 64, B), B, 0, stream>>>(r1w1, t_r11, 1296, 64);
  k_prep_w<<<cdiv(48 * 64, B), B, 0, stream>>>(r1wd, t_r1d, 48, 64);
  k_prep_w<<<cdiv(1728 * 64, B), B, 0, stream>>>(r1w2, t_r12, 1728, 64);
  k_prep_w<<<cdiv(1944 * 24, B), B, 0, stream>>>(u1w1, t_u11, 1944, 24);
  k_prep_w<<<cdiv(72 * 24, B), B, 0, stream>>>(u1wd, t_u1d, 72, 24);
  k_prep_w<<<cdiv(648 * 24, B), B, 0, stream>>>(u1w2, t_u12, 648, 24);
  k_prep_w<<<cdiv(48 * 64, B), B, 0, stream>>>(p0_w, t_p0, 48, 64);
  k_prep_w<<<cdiv(64 * 24, B), B, 0, stream>>>(p1_w, t_p1, 64, 24);

  // 1. point -> voxel scatter of counts and z_depth
  k_scatter_base<<<cdiv(N, B), B, 0, stream>>>(points, zdepth, pres, vres, cnt1, vd, N);
  k_finalize_base<<<cdiv(V32, B), B, 0, stream>>>(cnt1, vd, mask0, V32);
  k_pool<<<cdiv(V16, B), B, 0, stream>>>(mask0, cnt1, mask1, cnt2);
  k_maskmul_h<<<cdiv(V32 * 32, B), B, 0, stream>>>(featg, mask0, featm_h, V32, 32);

  // 2. stem: x0 = CBR(vd,1->48) (scalar stencil) + CBR(feat*mask,32->48) (WMMA, accum)
  k_stem_d<<<cdiv(V32 * 48, B), B, 0, stream>>>(vd, sd_w, sd_g, sd_b, sd_m, sd_v, mask0, x0);
  conv_wmma<_Float16, float, 3, 32, 48, 32, 32, 1, 1>
      <<<dim3(V32 / 16, 3), 32, 0, stream>>>(
      featm_h, t_sf, nullptr, sf_g, sf_b, sf_m, sf_v, mask0, nullptr, x0, zrowh, 1, 1);

  // 3. z0 = devox(x0, stride 1)
  k_devox<<<cdiv(N * 48, B), B, 0, stream>>>(x0, points, pres, vres, z0, N, 32, 48, 1, 0);

  // 4. x1 = CBR_s2(vox_mean(z0)*mask0) ; resblock at 16^3
  k_scatter_feat<<<cdiv(N, B), B, 0, stream>>>(points, z0, pres, vres, g32, N, 48, 0, 32);
  k_norm_mean_h<<<cdiv(V32 * 48, B), B, 0, stream>>>(g32, cnt1, mask0, g32h, V32, 48);
  conv_wmma<_Float16, _Float16, 2, 48, 48, 32, 16, 2, 0>
      <<<dim3(V16 / 16, 3), 32, 0, stream>>>(
      g32h, t_dn, nullptr, dn_g, dn_b, dn_m, dn_v, mask1, nullptr, x1a_h, zrowh, 1, 0);
  conv_wmma<_Float16, _Float16, 3, 48, 64, 16, 16, 1, 1>
      <<<dim3(V16 / 16, 4), 32, 0, stream>>>(
      x1a_h, t_r11, nullptr, r1g1, r1b1, r1m1, r1v1, mask1, nullptr, h1h, zrowh, 1, 0);
  conv_wmma<_Float16, float, 1, 48, 64, 16, 16, 1, 0>
      <<<dim3(V16 / 16, 4), 32, 0, stream>>>(
      x1a_h, t_r1d, nullptr, r1gd, r1bd, r1md, r1vd, nullptr, nullptr, s1, zrowh, 0, 0);
  conv_wmma<_Float16, float, 3, 64, 64, 16, 16, 1, 1>
      <<<dim3(V16 / 16, 4), 32, 0, stream>>>(
      h1h, t_r12, nullptr, r1g2, r1b2, r1m2, r1v2, mask1, s1, x1, zrowh, 1, 0);

  // 5. z1 = ptmlp0(z0) + devox(x1, stride 2)   (point mode: G=59, 59^3 >= N)
  conv_wmma<float, float, 1, 48, 64, 59, 59, 1, 0>
      <<<dim3(N / 16, 4), 32, 0, stream>>>(
      z0, t_p0, p0_b, p0g, p0b, p0m, p0v, nullptr, nullptr, z1, zrow, 1, 0);
  k_devox<<<cdiv(N * 64, B), B, 0, stream>>>(x1, points, pres, vres, z1, N, 16, 64, 2, 1);

  // 6. y2 = resblock32( concat( relu(bn(deconv(vox_mean(z1)*mask1)))*mask0 , x0 ) )
  k_scatter_feat<<<cdiv(N, B), B, 0, stream>>>(points, z1, pres, vres, g16, N, 64, 1, 16);
  k_norm_mean<<<cdiv(V16 * 64, B), B, 0, stream>>>(g16, cnt2, mask1, V16, 64);
  k_deconv<<<cdiv(V32 * 24, B), B, 0, stream>>>(g16, dc_w, dc_g, dc_b, dc_m, dc_v, mask0, y2cat_h);
  k_concat<<<cdiv(V32 * 48, B), B, 0, stream>>>(x0, y2cat_h, V32);
  conv_wmma<_Float16, _Float16, 3, 72, 24, 32, 32, 1, 1>
      <<<dim3(V32 / 16, 2), 32, 0, stream>>>(
      y2cat_h, t_u11, nullptr, u1g1, u1b1, u1m1, u1v1, mask0, nullptr, h2h, zrowh, 1, 0);
  conv_wmma<_Float16, float, 1, 72, 24, 32, 32, 1, 0>
      <<<dim3(V32 / 16, 2), 32, 0, stream>>>(
      y2cat_h, t_u1d, nullptr, u1gd, u1bd, u1md, u1vd, nullptr, nullptr, s2, zrowh, 0, 0);
  conv_wmma<_Float16, float, 3, 24, 24, 32, 32, 1, 1>
      <<<dim3(V32 / 16, 2), 32, 0, stream>>>(
      h2h, t_u12, nullptr, u1g2, u1b2, u1m2, u1v2, mask0, s2, y2f, zrowh, 1, 0);

  // 7. z2 = ptmlp1(z1) + devox(y2, stride 1);  out = z2 @ occ.w + occ.b
  conv_wmma<float, float, 1, 64, 24, 59, 59, 1, 0>
      <<<dim3(N / 16, 2), 32, 0, stream>>>(
      z1, t_p1, p1_b, p1g, p1b, p1m, p1v, nullptr, nullptr, z2, zrow, 1, 0);
  k_devox<<<cdiv(N * 24, B), B, 0, stream>>>(y2f, points, pres, vres, z2, N, 32, 24, 1, 1);
  k_occ<<<cdiv(N, B), B, 0, stream>>>(z2, oc_w, oc_b, (float*)d_out, N);
}
